// LSTM_Net_47880295415874
// MI455X (gfx1250) — matrix operand order repaired and checked by reading, not compiled, not run on hardware
//
#include <hip/hip_runtime.h>
#include <hip/hip_bf16.h>
#include <math.h>

// ---------------------------------------------------------------------------
// Problem constants (from reference): B=1024, T=128, H=128, D0=64, G=4H=512
// ---------------------------------------------------------------------------
#define BATCH 1024
#define TSEQ  128
#define HID   128
#define GATES 512   // 4*H
#define NT    32    // GATES/16 N-tiles
#define BT    16    // batch rows per workgroup (one WMMA M tile)
#define NTHR  256   // 8 wave32s

typedef __attribute__((ext_vector_type(16))) __bf16 v16bf;
typedef __attribute__((ext_vector_type(8)))  float  v8f;

union ABReg { v16bf v; uint4 q[2]; unsigned int u[8]; unsigned short s[16]; };

// ---- bf16 helpers ----------------------------------------------------------
__device__ __forceinline__ unsigned short to_bf(float f) {
    union { float f; unsigned u; } v; v.f = f;
    unsigned r = v.u + 0x7FFFu + ((v.u >> 16) & 1u);  // round-nearest-even
    return (unsigned short)(r >> 16);
}
__device__ __forceinline__ unsigned short to_bf(unsigned short s) { return s; }
__device__ __forceinline__ float bf_to_f(unsigned short h) {
    union { unsigned u; float f; } v; v.u = ((unsigned)h) << 16; return v.f;
}

// ---- fast gate activations (f32, transcendental VALU) ----------------------
__device__ __forceinline__ float fsig(float x) {
    return 1.0f / (1.0f + __expf(-x));
}
__device__ __forceinline__ float ftanh(float x) {
    float e = __expf(-2.0f * x);
    return (1.0f - e) / (1.0f + e);
}

// ---------------------------------------------------------------------------
// Pack a row-major f32 weight W[GATES][K] into the bf16 WMMA B-operand layout.
// B operand for one (kt,nt) tile: 32 lanes x 8 dwords (2 bf16 each) = 1KB.
//   lanes 0-15 : N = lane,    dword j holds K = kt*32 + 2j, 2j+1
//   lanes16-31 : N = lane-16, dword j holds K = kt*32 + 16 + 2j, 2j+1
// packed[] indexed in dwords: ((kt*NT + nt)*256 + lane*8 + j)
// ---------------------------------------------------------------------------
__device__ void pack_weight_bf16(const float* __restrict__ W,
                                 unsigned int* __restrict__ packed,
                                 int K, int tid) {
    int total = GATES * K;
    for (int e = tid; e < total; e += NTHR) {
        int g = e / K, k = e % K;
        int kt = k >> 5, kk = k & 31;
        int nt = g >> 4, n  = g & 15;
        int lane = (kk < 16) ? n : (16 + n);
        int j    = (kk & 15) >> 1;
        int half = kk & 1;
        unsigned short* dst =
            (unsigned short*)(packed + ((size_t)(kt * NT + nt) * 256 + lane * 8 + j));
        dst[half] = to_bf(W[e]);
    }
}

// ---------------------------------------------------------------------------
// Pack WMMA A operand (16x32 bf16) from a row-major bf16 LDS tile.
//   lanes 0-15 : M = lane,    v0-3: K=0..7,   v4-7: K=8..15
//   lanes16-31 : M = lane-16, v0-3: K=16..23, v4-7: K=24..31
// Both halves are 16 consecutive bytes, 16B aligned -> 2x ds_load_b128.
// ---------------------------------------------------------------------------
__device__ __forceinline__ void pack_A(ABReg& a,
                                       const unsigned short* __restrict__ tile,
                                       int stride, int kbase, int m, int hi) {
    const unsigned short* row = tile + m * stride + kbase + (hi ? 16 : 0);
    a.q[0] = *(const uint4*)(row);        // K = kbase(+16) .. +7
    a.q[1] = *(const uint4*)(row + 8);    // K = kbase+8(+16) .. +7
}

__device__ __forceinline__ void load_B(ABReg& b,
                                       const unsigned int* __restrict__ packed,
                                       int kt, int nt, int lane) {
    const uint4* p = (const uint4*)(packed + ((size_t)(kt * NT + nt) * 256 + lane * 8));
    b.q[0] = p[0];                        // -> ds_load_b128
    b.q[1] = p[1];
}

// ---------------------------------------------------------------------------
// Fused persistent LSTM layer kernel. One block = 16 batch rows for all T
// timesteps. 8 waves; wave w owns N-tiles {w, 8+w, 16+w, 24+w} = the same 16
// hidden columns of the i/f/g/o gate regions, so the cell update is wave-local.
// All weight B-operands are hoisted into VGPRs before the time loop
// (layer1: 32 operands x 8 VGPRs = 256 registers of resident weights).
// ---------------------------------------------------------------------------
template <typename InT, int DIN>
__global__ __launch_bounds__(NTHR)
void lstm_layer_kernel(const InT*  __restrict__ in,     // (B, T, DIN)
                       const float* __restrict__ Wih,   // (4H, DIN)
                       const float* __restrict__ Whh,   // (4H, H)
                       const float* __restrict__ bih,   // (4H)
                       const float* __restrict__ bhh,   // (4H)
                       unsigned short* __restrict__ hseq) // (B, T, H) bf16 out
{
    extern __shared__ unsigned char smem[];
    unsigned int*   pWih   = (unsigned int*)smem;                        // GATES*DIN*2 B
    unsigned int*   pWhh   = (unsigned int*)(smem + (size_t)GATES * DIN * 2);
    float*          biasS  = (float*)(smem + (size_t)GATES * DIN * 2 + (size_t)GATES * HID * 2);
    unsigned short* hTile  = (unsigned short*)((unsigned char*)biasS + GATES * 4); // 16 x H bf16
    unsigned short* xStage = hTile + BT * HID;                           // 16 x DIN bf16

    const int tid  = threadIdx.x;
    const int lane = tid & 31;
    const int w    = tid >> 5;           // wave id 0..7
    const int b0   = blockIdx.x * BT;    // first batch row of this tile
    const int hi   = (lane >= 16) ? 1 : 0;
    const int nloc = lane & 15;          // N within tile / M row selector
    const int col  = w * 16 + nloc;      // hidden column owned by this lane

    // ---- one-time LDS setup: pack weights (bf16 B-operand layout), bias, h=0
    pack_weight_bf16(Wih, pWih, DIN, tid);
    pack_weight_bf16(Whh, pWhh, HID, tid);
    for (int i = tid; i < GATES; i += NTHR) biasS[i] = bih[i] + bhh[i];
    for (int i = tid; i < BT * HID; i += NTHR) hTile[i] = 0;
    __syncthreads();

    // ---- hoist ALL weight B-operands into registers (loop-invariant) ----
    constexpr int KTI = DIN / 32;                    // input K-tiles (2 or 4)
    ABReg bIh[KTI][4];                               // [kt][gate]
    ABReg bHh[4][4];                                 // [kt][gate]
#pragma unroll
    for (int kt = 0; kt < KTI; ++kt)
#pragma unroll
        for (int gi = 0; gi < 4; ++gi)
            load_B(bIh[kt][gi], pWih, kt, gi * 8 + w, lane);
#pragma unroll
    for (int kt = 0; kt < 4; ++kt)
#pragma unroll
        for (int gi = 0; gi < 4; ++gi)
            load_B(bHh[kt][gi], pWhh, kt, gi * 8 + w, lane);

    // per-lane gate bias (same for all 8 accumulator rows)
    float bv[4];
#pragma unroll
    for (int gi = 0; gi < 4; ++gi) bv[gi] = biasS[gi * 128 + col];

    float cst[8];
#pragma unroll
    for (int j = 0; j < 8; ++j) cst[j] = 0.0f;

    for (int t = 0; t < TSEQ; ++t) {
        // ---- stage x_t tile (16 x DIN) into LDS as bf16 ----
        for (int i = tid; i < BT * DIN; i += NTHR) {
            int r = i / DIN, d = i % DIN;
            xStage[i] = to_bf(in[((size_t)(b0 + r) * TSEQ + t) * DIN + d]);
        }
        // prefetch next timestep's rows (global_prefetch_b8)
        if (t + 1 < TSEQ && tid < BT)
            __builtin_prefetch(&in[((size_t)(b0 + tid) * TSEQ + (t + 1)) * DIN], 0, 1);
        __syncthreads();   // x_t staged; h from previous step visible

        // ---- gates = x_t @ Wih^T + h @ Whh^T (bias folded into epilogue)
        ABReg a;
        v8f acc[4];

        pack_A(a, xStage, DIN, 0, nloc, hi);
#pragma unroll
        for (int gi = 0; gi < 4; ++gi) {
            v8f zc = {};                      // inline-0 C operand
            acc[gi] = __builtin_amdgcn_wmma_f32_16x16x32_bf16(
                false, a.v, false, bIh[0][gi].v, (short)0, zc, false, false);
        }
#pragma unroll
        for (int kt = 1; kt < KTI; ++kt) {
            pack_A(a, xStage, DIN, kt * 32, nloc, hi);
#pragma unroll
            for (int gi = 0; gi < 4; ++gi)
                acc[gi] = __builtin_amdgcn_wmma_f32_16x16x32_bf16(
                    false, a.v, false, bIh[kt][gi].v, (short)0, acc[gi], false, false);
        }
#pragma unroll
        for (int kt = 0; kt < 4; ++kt) {      // recurrent GEMM
            pack_A(a, hTile, HID, kt * 32, nloc, hi);
#pragma unroll
            for (int gi = 0; gi < 4; ++gi)
                acc[gi] = __builtin_amdgcn_wmma_f32_16x16x32_bf16(
                    false, a.v, false, bHh[kt][gi].v, (short)0, acc[gi], false, false);
        }
        __syncthreads();   // all waves done reading hTile

        // ---- gate nonlinearities + cell/hidden update (f32 VALU) ----
        // C/D layout: lanes0-15 vgpr j -> M=j ; lanes16-31 -> M=8+j ; N=lane&15
#pragma unroll
        for (int j = 0; j < 8; ++j) {
            float gi_ = fsig (acc[0][j] + bv[0]);
            float gf_ = fsig (acc[1][j] + bv[1]);
            float gg_ = ftanh(acc[2][j] + bv[2]);
            float go_ = fsig (acc[3][j] + bv[3]);
            float c   = gf_ * cst[j] + gi_ * gg_;
            cst[j]    = c;
            float h   = go_ * ftanh(c);
            unsigned short hb = to_bf(h);
            int m = j + (hi ? 8 : 0);
            hTile[m * HID + col] = hb;                                   // LDS state
            hseq[((size_t)(b0 + m) * TSEQ + t) * HID + col] = hb;        // bf16 seq out
        }
        // next iteration's first __syncthreads() makes hTile writes visible
    }
}

// ---------------------------------------------------------------------------
// Final FC: out(B,3) = h2[:, T-1, :] @ Wfc^T + bfc
// ---------------------------------------------------------------------------
__global__ __launch_bounds__(128)
void fc_kernel(const unsigned short* __restrict__ h2,  // (B,T,H) bf16
               const float* __restrict__ Wfc,          // (3,H)
               const float* __restrict__ bfc,          // (3)
               float* __restrict__ out)                // (B,3)
{
    __shared__ float red[128];
    const int b = blockIdx.x, tid = threadIdx.x;
    float h = bf_to_f(h2[((size_t)b * TSEQ + (TSEQ - 1)) * HID + tid]);
#pragma unroll
    for (int j = 0; j < 3; ++j) {
        red[tid] = h * Wfc[j * HID + tid];
        __syncthreads();
        for (int s = 64; s > 0; s >>= 1) {
            if (tid < s) red[tid] += red[tid + s];
            __syncthreads();
        }
        if (tid == 0) out[b * 3 + j] = red[0] + bfc[j];
        __syncthreads();
    }
}

// ---------------------------------------------------------------------------
extern "C" void kernel_launch(void* const* d_in, const int* in_sizes, int n_in,
                              void* d_out, int out_size, void* d_ws, size_t ws_size,
                              hipStream_t stream) {
    const float* x    = (const float*)d_in[0];
    const float* Wih0 = (const float*)d_in[1];
    const float* Whh0 = (const float*)d_in[2];
    const float* bih0 = (const float*)d_in[3];
    const float* bhh0 = (const float*)d_in[4];
    const float* Wih1 = (const float*)d_in[5];
    const float* Whh1 = (const float*)d_in[6];
    const float* bih1 = (const float*)d_in[7];
    const float* bhh1 = (const float*)d_in[8];
    const float* Wfc  = (const float*)d_in[9];
    const float* bfc  = (const float*)d_in[10];
    float* out = (float*)d_out;

    // workspace: h1 seq (bf16) | h2 seq (bf16), 32MB each
    const size_t seq_elems = (size_t)BATCH * TSEQ * HID;
    unsigned short* h1 = (unsigned short*)d_ws;
    unsigned short* h2 = h1 + seq_elems;

    // dynamic LDS: packed Wih + packed Whh + bias + h tile + x stage
    const size_t lds0 = (size_t)GATES * 64  * 2 + (size_t)GATES * HID * 2
                      + GATES * 4 + BT * HID * 2 + BT * 64  * 2;   // ~200 KB
    const size_t lds1 = (size_t)GATES * HID * 2 + (size_t)GATES * HID * 2
                      + GATES * 4 + BT * HID * 2 + BT * HID * 2;   // ~266 KB

    dim3 grid(BATCH / BT);   // 64 workgroups, each persistent over T
    dim3 block(NTHR);

    lstm_layer_kernel<float, 64>
        <<<grid, block, lds0, stream>>>(x,  Wih0, Whh0, bih0, bhh0, h1);
    lstm_layer_kernel<unsigned short, 128>
        <<<grid, block, lds1, stream>>>(h1, Wih1, Whh1, bih1, bhh1, h2);
    fc_kernel<<<dim3(BATCH), dim3(128), 0, stream>>>(h2, Wfc, bfc, out);
}
